// NodeEmbedder2_71150428226104
// MI455X (gfx1250) — matrix operand order repaired and checked by reading, not compiled, not run on hardware
//
#include <hip/hip_runtime.h>
#include <hip/hip_bf16.h>
#include <math.h>

// ---------------------------------------------------------------------------
// Types for CDNA5 WMMA
// ---------------------------------------------------------------------------
typedef __bf16 bf16_t;
typedef bf16_t v16bf __attribute__((ext_vector_type(16)));
typedef float  v8f   __attribute__((ext_vector_type(8)));

#define HDIM   128
#define FDIM   32
#define NCONV  3
#define CATW   (FDIM + (NCONV + 1) * HDIM)   // 544

// pack two floats (consecutive K: lo = even k, hi = odd k) into one bf16x2
// dword; __float22bfloat162_rn lowers to the packed v_cvt_pk_bf16_f32 path
static __device__ __forceinline__ unsigned pack_bf2(float lo, float hi) {
    __hip_bfloat162 p = __float22bfloat162_rn(float2{lo, hi});
    return *(unsigned*)&p;
}

// ---------------------------------------------------------------------------
// Stage one 128x32 A chunk (row-major) and 32x128 B chunk (transposed) into
// LDS as bf16. Branch-free: out-of-range M rows are clamped (garbage only
// reaches C rows that are never stored).
// ---------------------------------------------------------------------------
static __device__ __forceinline__ void stage_tile(
    const float* __restrict__ A, int lda,
    const float* __restrict__ B,           // ldb == 128
    int M, int rowBase, int kc, int t,
    unsigned short* __restrict__ sA, unsigned short* __restrict__ sB)
{
    #pragma unroll
    for (int j = 0; j < 4; ++j) {
        int idx = t + j * 256;             // (row, kquad): 128 x 8
        int r  = idx >> 3;
        int kq = idx & 7;
        int row = rowBase + r;
        row = (row < M) ? row : (M - 1);   // clamp: branch-free
        float4 f = *(const float4*)(A + (long)row * lda + kc + kq * 4);
        unsigned* dstp = (unsigned*)&sA[r * 32 + kq * 4];
        dstp[0] = pack_bf2(f.x, f.y);
        dstp[1] = pack_bf2(f.z, f.w);
    }
    #pragma unroll
    for (int j = 0; j < 2; ++j) {
        int idx = t + j * 256;             // 16 kpairs x 32 nquads = 512
        int kp = idx >> 5;
        int nq = idx & 31;
        int k = kp * 2, n = nq * 4;
        const float* bp = B + (long)(kc + k) * 128 + n;
        float4 f0 = *(const float4*)bp;            // row k
        float4 f1 = *(const float4*)(bp + 128);    // row k+1
        *(unsigned*)&sB[(n + 0) * 32 + k] = pack_bf2(f0.x, f1.x);
        *(unsigned*)&sB[(n + 1) * 32 + k] = pack_bf2(f0.y, f1.y);
        *(unsigned*)&sB[(n + 2) * 32 + k] = pack_bf2(f0.z, f1.z);
        *(unsigned*)&sB[(n + 3) * 32 + k] = pack_bf2(f0.w, f1.w);
    }
}

// ---------------------------------------------------------------------------
// GEMM: C[M x 128] = act(A[M x K] @ B[K x 128] + bias)
// BM=128, BN=128, BK=32; 256 threads = 8 waves arranged 4(M) x 2(N),
// each wave owns a 32x64 tile -> 8 v_wmma_f32_16x16x32_bf16 per K-chunk.
// Double-buffered LDS: one barrier per chunk; next chunk's global loads
// issue before the current chunk's WMMAs (latency hiding).
// ---------------------------------------------------------------------------
template <bool BIAS, bool RELU>
__global__ __launch_bounds__(256)
void gemm_bf16_wmma(const float* __restrict__ A, int lda,
                    const float* __restrict__ B,       // ldb == 128, row-major KxN
                    const float* __restrict__ bias,
                    float* __restrict__ C, int ldc,
                    int M, int K)
{
    __shared__ unsigned short sA[2][128 * 32];  // bf16, row-major chunks
    __shared__ unsigned short sB[2][128 * 32];  // bf16, transposed [n*32+k]

    const int t    = threadIdx.x;
    const int lane = t & 31;
    const int wave = t >> 5;
    const int wm   = wave >> 1;      // 0..3  (M tile, 32 rows each)
    const int wn   = wave & 1;       // 0..1  (N tile, 64 cols each)
    const int rowBase = blockIdx.x * 128;
    const int hi  = lane >> 4;       // half-wave select
    const int l16 = lane & 15;

    v8f zero = {0.f,0.f,0.f,0.f,0.f,0.f,0.f,0.f};
    v8f acc[2][4];
    #pragma unroll
    for (int i = 0; i < 2; ++i)
        #pragma unroll
        for (int j = 0; j < 4; ++j)
            acc[i][j] = zero;

    const int nch = K >> 5;          // K is always a multiple of 32 here
    stage_tile(A, lda, B, M, rowBase, 0, t, sA[0], sB[0]);

    for (int ch = 0; ch < nch; ++ch) {
        __syncthreads();             // staged buffer [ch&1] is ready;
                                     // previous compute done before overwrite
        if (ch + 1 < nch)
            stage_tile(A, lda, B, M, rowBase, (ch + 1) * 32, t,
                       sA[(ch + 1) & 1], sB[(ch + 1) & 1]);

        const unsigned short* cA = sA[ch & 1];
        const unsigned short* cB = sB[ch & 1];

        union Frag { v16bf v; unsigned u[8]; };
        Frag fa[2], fb[4];
        // A fragments: lane holds row ml; K pairs per ISA 16-bit A layout
        #pragma unroll
        for (int s = 0; s < 2; ++s) {
            int ml = wm * 32 + s * 16 + l16;
            #pragma unroll
            for (int v = 0; v < 8; ++v) {
                int kk = (v < 4) ? (hi * 8 + v * 2) : (16 + hi * 8 + (v - 4) * 2);
                fa[s].u[v] = *(const unsigned*)&cA[ml * 32 + kk];
            }
        }
        // B fragments: lane holds column nl; K pairs in the same order as A
        #pragma unroll
        for (int s = 0; s < 4; ++s) {
            int nl = wn * 64 + s * 16 + l16;
            #pragma unroll
            for (int v = 0; v < 8; ++v) {
                int kk = (v < 4) ? (hi * 8 + v * 2) : (16 + hi * 8 + (v - 4) * 2);
                fb[s].u[v] = *(const unsigned*)&cB[nl * 32 + kk];
            }
        }
        #pragma unroll
        for (int i = 0; i < 2; ++i)
            #pragma unroll
            for (int j = 0; j < 4; ++j)
                acc[i][j] = __builtin_amdgcn_wmma_f32_16x16x32_bf16(
                    false, fa[i].v, false, fb[j].v, (short)0, acc[i][j],
                    false, false);
    }

    // ---- epilogue: ISA C layout -> lane = col l16; VGPR r = row r + 8*hi ----
    float bz[4];
    #pragma unroll
    for (int j = 0; j < 4; ++j)
        bz[j] = BIAS ? bias[wn * 64 + j * 16 + l16] : 0.0f;

    const int rbase = rowBase + wm * 32 + hi * 8;
    if (rowBase + 128 <= M) {
        // fast path: full tile, no per-element guards (781 of 782 blocks)
        #pragma unroll
        for (int i = 0; i < 2; ++i) {
            float* cp = C + (size_t)(rbase + i * 16) * ldc + wn * 64 + l16;
            #pragma unroll
            for (int j = 0; j < 4; ++j)
                #pragma unroll
                for (int r = 0; r < 8; ++r) {
                    float v = acc[i][j][r] + bz[j];
                    if (RELU) v = fmaxf(v, 0.0f);
                    cp[(size_t)r * ldc + j * 16] = v;
                }
        }
    } else {
        // edge block: guarded stores
        #pragma unroll
        for (int i = 0; i < 2; ++i) {
            #pragma unroll
            for (int j = 0; j < 4; ++j) {
                int col = wn * 64 + j * 16 + l16;
                #pragma unroll
                for (int r = 0; r < 8; ++r) {
                    int row = rbase + i * 16 + r;
                    if (row < M) {
                        float v = acc[i][j][r] + bz[j];
                        if (RELU) v = fmaxf(v, 0.0f);
                        C[(size_t)row * ldc + col] = v;
                    }
                }
            }
        }
    }
}

// ---------------------------------------------------------------------------
// cat[:, 0:32] = x
// ---------------------------------------------------------------------------
__global__ __launch_bounds__(256)
void copy_x_kernel(const float* __restrict__ x, float* __restrict__ cat, int n)
{
    int i = blockIdx.x * 256 + threadIdx.x;
    if (i < n * FDIM) {
        int r = i >> 5, c = i & 31;
        cat[(long)r * CATW + c] = x[i];
    }
}

// ---------------------------------------------------------------------------
// per-node attention logits: a_s[n]=dot(xh[n],att_s), a_d[n]=dot(xh[n],att_d)
// one wave32 per node; float4 per lane; shuffle reduction
// ---------------------------------------------------------------------------
__global__ __launch_bounds__(256)
void node_att_kernel(const float* __restrict__ xh,
                     const float* __restrict__ att_s,
                     const float* __restrict__ att_d,
                     float* __restrict__ as_, float* __restrict__ ad_, int n)
{
    int node = (blockIdx.x * 256 + threadIdx.x) >> 5;
    int lane = threadIdx.x & 31;
    if (node >= n) return;
    float4 v = ((const float4*)(xh + (long)node * HDIM))[lane];
    float4 a = ((const float4*)att_s)[lane];
    float4 b = ((const float4*)att_d)[lane];
    float ss = v.x*a.x + v.y*a.y + v.z*a.z + v.w*a.w;
    float dd = v.x*b.x + v.y*b.y + v.z*b.z + v.w*b.w;
    for (int o = 16; o > 0; o >>= 1) {
        ss += __shfl_down(ss, o, 32);
        dd += __shfl_down(dd, o, 32);
    }
    if (lane == 0) { as_[node] = ss; ad_[node] = dd; }
}

// ---------------------------------------------------------------------------
// init conv accumulators: outblk[n][c] = bias[c]; amax = -inf; denom = 0
// ---------------------------------------------------------------------------
__global__ __launch_bounds__(256)
void init_conv_kernel(float* __restrict__ outblk, const float* __restrict__ bias,
                      float* __restrict__ amax, float* __restrict__ denom, int n)
{
    int i = blockIdx.x * 256 + threadIdx.x;
    if (i < n * HDIM) {
        int r = i >> 7, c = i & 127;
        outblk[(long)r * CATW + c] = bias[c];
        if (c == 0) {
            amax[r]  = __uint_as_float(0xFF800000u);  // -inf
            denom[r] = 0.0f;
        }
    }
}

// ---------------------------------------------------------------------------
// edge phase 1: alpha = leaky_relu(a_s[src]+a_d[dst]); segment max into amax
// float atomic-max via int/uint bit trick (global_atomic_max_i32 / min_u32)
// ---------------------------------------------------------------------------
__global__ __launch_bounds__(256)
void edge_alpha_kernel(const int* __restrict__ src, const int* __restrict__ dst,
                       const float* __restrict__ as_, const float* __restrict__ ad_,
                       float* __restrict__ alpha, float* __restrict__ amax,
                       int E, int n)
{
    int e = blockIdx.x * 256 + threadIdx.x;
    int Ep = E + n;
    if (e >= Ep) return;
    int s = (e < E) ? src[e] : (e - E);   // appended self-loops
    int d = (e < E) ? dst[e] : (e - E);
    float al = as_[s] + ad_[d];
    al = (al > 0.0f) ? al : 0.2f * al;    // leaky_relu, slope 0.2
    alpha[e] = al;
    if (al >= 0.0f) atomicMax((int*)(amax + d), __float_as_int(al));
    else            atomicMin((unsigned int*)(amax + d), __float_as_uint(al));
}

// ---------------------------------------------------------------------------
// edge phase 2: ex = exp(alpha - amax[dst]) in place; denom[dst] += ex
// ---------------------------------------------------------------------------
__global__ __launch_bounds__(256)
void edge_exp_kernel(const int* __restrict__ dst,
                     float* __restrict__ alpha, const float* __restrict__ amax,
                     float* __restrict__ denom, int E, int n)
{
    int e = blockIdx.x * 256 + threadIdx.x;
    int Ep = E + n;
    if (e >= Ep) return;
    int d = (e < E) ? dst[e] : (e - E);
    float ex = __expf(alpha[e] - amax[d]);
    alpha[e] = ex;
    atomicAdd(denom + d, ex);
}

// ---------------------------------------------------------------------------
// edge phase 3: out[dst] += xh[src] * (ex / denom[dst]); one wave32 per edge,
// float4 per lane across the 128 channels, f32 atomics (resolve in L2)
// ---------------------------------------------------------------------------
__global__ __launch_bounds__(256)
void edge_agg_kernel(const int* __restrict__ src, const int* __restrict__ dst,
                     const float* __restrict__ alpha, const float* __restrict__ denom,
                     const float* __restrict__ xh, float* __restrict__ outblk,
                     int E, int n)
{
    int e = (blockIdx.x * 256 + threadIdx.x) >> 5;
    int lane = threadIdx.x & 31;
    int Ep = E + n;
    if (e >= Ep) return;
    int s = (e < E) ? src[e] : (e - E);
    int d = (e < E) ? dst[e] : (e - E);
    float w = alpha[e] / (denom[d] + 1e-16f);
    float4 v = ((const float4*)(xh + (long)s * HDIM))[lane];
    float* o = outblk + (long)d * CATW + lane * 4;
    atomicAdd(o + 0, v.x * w);
    atomicAdd(o + 1, v.y * w);
    atomicAdd(o + 2, v.z * w);
    atomicAdd(o + 3, v.w * w);
}

// ---------------------------------------------------------------------------
// launch
// ---------------------------------------------------------------------------
extern "C" void kernel_launch(void* const* d_in, const int* in_sizes, int n_in,
                              void* d_out, int out_size, void* d_ws, size_t ws_size,
                              hipStream_t stream)
{
    const float* x    = (const float*)d_in[0];
    const int*   src  = (const int*)  d_in[1];
    const int*   dst  = (const int*)  d_in[2];
    const float* pw1  = (const float*)d_in[3];
    const float* pb1  = (const float*)d_in[4];
    const float* pw2  = (const float*)d_in[5];
    const float* pb2  = (const float*)d_in[6];
    const float* gw   = (const float*)d_in[7];   // [NC,128,128]
    const float* gas  = (const float*)d_in[8];   // [NC,1,128]
    const float* gad  = (const float*)d_in[9];   // [NC,1,128]
    const float* gb   = (const float*)d_in[10];  // [NC,128]
    const float* mw1  = (const float*)d_in[11];  // [544,128]
    const float* mb1  = (const float*)d_in[12];
    const float* mw2  = (const float*)d_in[13];  // [128,128]
    const float* mb2  = (const float*)d_in[14];

    const int N = in_sizes[0] / FDIM;
    const int E = in_sizes[1];
    const int Ep = E + N;

    // workspace partition (floats)
    float* ws    = (float*)d_ws;
    float* cat   = ws;                             // N x 544 (jumping-knowledge)
    float* xh    = cat   + (size_t)N * CATW;       // N x 128
    float* hid   = xh    + (size_t)N * HDIM;       // N x 128
    float* as_   = hid   + (size_t)N * HDIM;       // N
    float* ad_   = as_   + (size_t)N;              // N
    float* amax  = ad_   + (size_t)N;              // N
    float* denom = amax  + (size_t)N;              // N
    float* alpha = denom + (size_t)N;              // E + N

    const int gemmGrid = (N + 127) / 128;
    const int nThr256  = 256;

    // cat[:,0:32] = x
    copy_x_kernel<<<(N * FDIM + 255) / 256, nThr256, 0, stream>>>(x, cat, N);

    // preprocess MLP: hid = relu(x @ pw1 + pb1); cat[:,32:160] = hid @ pw2 + pb2
    gemm_bf16_wmma<true, true><<<gemmGrid, nThr256, 0, stream>>>(
        x, FDIM, pw1, pb1, hid, HDIM, N, FDIM);
    gemm_bf16_wmma<true, false><<<gemmGrid, nThr256, 0, stream>>>(
        hid, HDIM, pw2, pb2, cat + FDIM, CATW, N, HDIM);

    // GATConv layers
    for (int i = 0; i < NCONV; ++i) {
        const float* hin  = (i == 0) ? (cat + FDIM) : (cat + FDIM + HDIM + (size_t)HDIM * (i - 1));
        float*       outb = cat + FDIM + HDIM + (size_t)HDIM * i;

        // xh = hin @ W_i  (no bias/act)
        gemm_bf16_wmma<false, false><<<gemmGrid, nThr256, 0, stream>>>(
            hin, CATW, gw + (size_t)i * HDIM * HDIM, nullptr, xh, HDIM, N, HDIM);

        node_att_kernel<<<(N * 32 + 255) / 256, nThr256, 0, stream>>>(
            xh, gas + (size_t)i * HDIM, gad + (size_t)i * HDIM, as_, ad_, N);

        init_conv_kernel<<<(N * HDIM + 255) / 256, nThr256, 0, stream>>>(
            outb, gb + (size_t)i * HDIM, amax, denom, N);

        edge_alpha_kernel<<<(Ep + 255) / 256, nThr256, 0, stream>>>(
            src, dst, as_, ad_, alpha, amax, E, N);

        edge_exp_kernel<<<(Ep + 255) / 256, nThr256, 0, stream>>>(
            dst, alpha, amax, denom, E, N);

        edge_agg_kernel<<<((size_t)Ep * 32 + 255) / 256, nThr256, 0, stream>>>(
            src, dst, alpha, denom, xh, outb, E, N);
    }

    // post MLP on the concat: hid = relu(cat @ mw1 + mb1); out = hid @ mw2 + mb2
    gemm_bf16_wmma<true, true><<<gemmGrid, nThr256, 0, stream>>>(
        cat, CATW, mw1, mb1, hid, HDIM, N, CATW);
    gemm_bf16_wmma<true, false><<<gemmGrid, nThr256, 0, stream>>>(
        hid, HDIM, mw2, mb2, (float*)d_out, HDIM, N, HDIM);
}
